// SelfAttention_90220083020243
// MI455X (gfx1250) — matrix operand order repaired and checked by reading, not compiled, not run on hardware
//
#include <hip/hip_runtime.h>
#include <hip/hip_bf16.h>

// ---------------------------------------------------------------------------
// Self-attention (B=4, C=256, H=W=64, R=32), flash-attention style, bf16 WMMA.
//   q = Wq@x, k = Wk@x, v = Wv@x  (1x1 convs = channel GEMMs)
//   out = scale * (V @ softmax(QK^T)^T) + x
// CDNA5 paths used: v_wmma_f32_16x16x32_bf16 for all GEMMs,
// global_load_async_to_lds_b128 + s_wait_asynccnt for K/V tile staging,
// ds_load_b128 fragment reads, s_wait_dscnt for the P relayout.
// The N x N attention matrix is never materialized (~536 MB HBM saved).
// ---------------------------------------------------------------------------

typedef unsigned int   u32;
typedef unsigned short u16;
typedef __attribute__((ext_vector_type(4)))  u32    v4u;
typedef __attribute__((ext_vector_type(8)))  u32    v8u;
typedef __attribute__((ext_vector_type(16))) __bf16 v16bf;
typedef __attribute__((ext_vector_type(8)))  float  v8f;

static constexpr int Bn = 4;     // batch
static constexpr int Cc = 256;   // channels
static constexpr int Rr = 32;    // reduced channels
static constexpr int Nn = 4096;  // H*W
static constexpr int LDV = 40;   // padded LDS row pitch (elems): 80B = 5x16B chunks

static __device__ __forceinline__ u16 f2bf(float f) {
  u32 u = __float_as_uint(f);
  u32 r = 0x7FFFu + ((u >> 16) & 1u);   // round-to-nearest-even
  return (u16)((u + r) >> 16);
}

// ---- CDNA5 async copy: global -> LDS, tracked by ASYNCcnt ------------------
static __device__ __forceinline__ void async_b128(u32 lds_off, const void* g) {
  asm volatile("global_load_async_to_lds_b128 %0, %1, off"
               :: "v"(lds_off), "v"((unsigned long long)(uintptr_t)g)
               : "memory");
}
static __device__ __forceinline__ void wait_async_le9() {
  asm volatile("s_wait_asynccnt 0x9" ::: "memory");
}
static __device__ __forceinline__ void wait_async_0() {
  asm volatile("s_wait_asynccnt 0x0" ::: "memory");
}
// low 32 bits of a generic pointer to __shared__ = LDS byte offset (aperture hi in [63:32])
static __device__ __forceinline__ u32 lds_off32(const void* p) {
  return (u32)(uintptr_t)p;
}

// ---- WMMA fragment loaders (CDNA5 wave32 layouts, CDNA5 ISA) ----
// A (16x32 bf16, M x K): lanes 0-15 -> M=lane, K {0..7,16..23};
//                        lanes 16-31 -> M=lane-16, K {8..15,24..31}.
static __device__ __forceinline__ v16bf load_A16x32(const u16* base, int ld) {
  const int lane = threadIdx.x & 31;
  const int row  = lane & 15;
  const int kh   = (lane >> 4) << 3;                 // 0 or 8
  const u16* p = base + (size_t)row * ld;
  v4u c0 = *(const v4u*)(p + kh);                    // K = kh .. kh+7
  v4u c1 = *(const v4u*)(p + 16 + kh);               // K = 16+kh .. 16+kh+7
  v8u r;
  r[0] = c0[0]; r[1] = c0[1]; r[2] = c0[2]; r[3] = c0[3];
  r[4] = c1[0]; r[5] = c1[1]; r[6] = c1[2]; r[7] = c1[3];
  return __builtin_bit_cast(v16bf, r);
}

// B (32x16 bf16, K x N), sourced from a row-major matrix whose ROWS are the
// B-matrix COLUMNS (i.e. B[k,col] = SRC[col*ld + k]):
//   lanes 0-15 -> col=lane,    K {0..7,16..23}
//   lanes 16-31 -> col=lane-16, K {8..15,24..31}
static __device__ __forceinline__ v16bf load_B32x16(const u16* base, int ld) {
  const int lane = threadIdx.x & 31;
  const int colr = lane & 15;
  const int k0   = (lane >> 4) << 3;                 // 0 or 8
  const u16* p = base + (size_t)colr * ld + k0;
  v4u c0 = *(const v4u*)p;
  v4u c1 = *(const v4u*)(p + 16);
  v8u r;
  r[0] = c0[0]; r[1] = c0[1]; r[2] = c0[2]; r[3] = c0[3];
  r[4] = c1[0]; r[5] = c1[1]; r[6] = c1[2]; r[7] = c1[3];
  return __builtin_bit_cast(v16bf, r);
}

static __device__ __forceinline__ v8f wmma_bf16(v16bf a, v16bf b, v8f c) {
  return __builtin_amdgcn_wmma_f32_16x16x32_bf16(
      /*neg_a=*/false, a, /*neg_b=*/false, b,
      /*c_mod=*/(short)0, c, /*reuse_a=*/false, /*reuse_b=*/false);
}

// ---------------------------------------------------------------------------
// Kernel 0a: transpose-convert x (C,N) f32 -> Xh (N,C) bf16, tiled via LDS.
// ---------------------------------------------------------------------------
__global__ __launch_bounds__(256) void transpose_cvt_kernel(
    const float* __restrict__ x, u16* __restrict__ Xh) {
  __shared__ float tile[32][33];
  const int b  = blockIdx.z;
  const int n0 = blockIdx.x * 32;
  const int c0 = blockIdx.y * 32;
  const float* xb = x + (size_t)b * Cc * Nn;
  u16* Xb = Xh + (size_t)b * Nn * Cc;
  for (int j = threadIdx.y; j < 32; j += 8)
    tile[j][threadIdx.x] = xb[(size_t)(c0 + j) * Nn + n0 + threadIdx.x];
  __syncthreads();
  for (int j = threadIdx.y; j < 32; j += 8)
    Xb[(size_t)(n0 + j) * Cc + c0 + threadIdx.x] = f2bf(tile[threadIdx.x][j]);
}

// ---------------------------------------------------------------------------
// Kernel 0b: flat f32 -> bf16 conversion (weights).
// ---------------------------------------------------------------------------
__global__ __launch_bounds__(256) void cvt_bf16_kernel(
    const float* __restrict__ src, u16* __restrict__ dst, int n) {
  int i = blockIdx.x * blockDim.x + threadIdx.x;
  if (i < n) dst[i] = f2bf(src[i]);
}

// ---------------------------------------------------------------------------
// Kernel 1: batched NT GEMM, bf16 in / bf16 out, fp32 accumulate.
//   Out[i,j] = sum_k A[i*lda + k] * Bm[j*ldb + k]
// One wave per 16x16 output tile; K stepped in 32s via WMMA.
// ---------------------------------------------------------------------------
__global__ __launch_bounds__(128) void gemm_nt_bf16_kernel(
    const u16* __restrict__ A, long long strideA,
    const u16* __restrict__ Bm, long long strideB,
    u16* __restrict__ Out, long long strideO,
    int M, int Nout, int Kd, int lda, int ldb, int ldo) {
  const int wavesPerBlock = blockDim.x >> 5;
  const int wave = threadIdx.x >> 5;
  const int ntiles = (M / 16) * (Nout / 16);
  const int tid = blockIdx.x * wavesPerBlock + wave;
  if (tid >= ntiles) return;                 // wave-uniform: EXEC stays all-1s
  const int b  = blockIdx.y;
  const int nt = Nout / 16;
  const int it = tid / nt;
  const int jt = tid % nt;

  const u16* Ab = A  + (size_t)b * strideA + (size_t)it * 16 * lda;
  const u16* Bb = Bm + (size_t)b * strideB + (size_t)jt * 16 * ldb;

  v8f acc = {0.f, 0.f, 0.f, 0.f, 0.f, 0.f, 0.f, 0.f};
  for (int k = 0; k < Kd; k += 32) {
    v16bf af = load_A16x32(Ab + k, lda);
    v16bf bf = load_B32x16(Bb + k, ldb);
    acc = wmma_bf16(af, bf, acc);
  }

  // C/D layout: VGPR i -> row (lane>=16 ? 8+i : i), col = lane&15
  const int lane = threadIdx.x & 31;
  const int colc = lane & 15;
  const int row0 = (lane >> 4) * 8;
  u16* Ob = Out + (size_t)b * strideO + (size_t)it * 16 * ldo + jt * 16;
#pragma unroll
  for (int i = 0; i < 8; ++i)
    Ob[(size_t)(row0 + i) * ldo + colc] = f2bf(acc[i]);
}

// ---------------------------------------------------------------------------
// Kernel 2: fused flash attention + epilogue.
//   Qh,Kh: (B, N, 32) bf16 row-major.  Vh: (B, C, N) bf16 row-major (= v[o,m]).
// Block = 128 threads (4 waves); 64 query rows per block, 16 per wave.
// Key tiles of 32 are double-buffered in LDS via async B128 copies
// (9 async ops per thread per tile: 1 for K [2KB], 8 for V [16KB]).
// ---------------------------------------------------------------------------
__global__ __launch_bounds__(128) void flash_attn_kernel(
    const u16* __restrict__ Qh, const u16* __restrict__ Kh,
    const u16* __restrict__ Vh, const float* __restrict__ x,
    const float* __restrict__ scale, float* __restrict__ out) {
  __shared__ __attribute__((aligned(16))) u16 KtLds[2][32 * LDV];   // 2 x 2.5 KB
  __shared__ __attribute__((aligned(16))) u16 VtLds[2][256 * LDV];  // 2 x 20 KB
  __shared__ __attribute__((aligned(16))) u16 Plds[4 * 16 * 32];    // 4 KB

  const int b    = blockIdx.y;
  const int tid  = threadIdx.x;          // 0..127
  const int wave = tid >> 5;
  const int lane = tid & 31;
  const int colc = lane & 15;
  const int row0 = (lane >> 4) * 8;
  const int qbase = blockIdx.x * 64 + wave * 16;

  const u16* Qb = Qh + ((size_t)b * Nn + qbase) * Rr;
  const u16* Kb = Kh + (size_t)b * Nn * Rr;
  const u16* Vb = Vh + (size_t)b * Cc * Nn;

  // Issue one key tile's async copies (K: 128 chunks, V: 1024 chunks of 16B).
  auto issue_tile = [&](int mb, int buf) {
    {  // K tile: row = 32 keys x 64B
      const int row = tid >> 2, c = tid & 3;
      async_b128(lds_off32(&KtLds[buf][0]) + (u32)(row * (LDV * 2) + c * 16),
                 (const char*)(Kb + (size_t)(mb + row) * Rr) + c * 16);
    }
#pragma unroll
    for (int i = 0; i < 8; ++i) {  // V tile: 256 rows (o) x 64B
      const int ch = tid + i * 128;
      const int row = ch >> 2, c = ch & 3;
      async_b128(lds_off32(&VtLds[buf][0]) + (u32)(row * (LDV * 2) + c * 16),
                 (const char*)(Vb + (size_t)row * Nn + mb) + c * 16);
    }
  };

  const v16bf qf = load_A16x32(Qb, Rr);   // resident Q fragment (16 x 32)

  v8f accO[16];
#pragma unroll
  for (int t = 0; t < 16; ++t)
    accO[t] = (v8f){0.f, 0.f, 0.f, 0.f, 0.f, 0.f, 0.f, 0.f};

  float m_i[8], l_i[8];
#pragma unroll
  for (int i = 0; i < 8; ++i) { m_i[i] = -1e30f; l_i[i] = 0.f; }

  u16* pw = Plds + wave * (16 * 32);
  const v8f zero8 = {0.f, 0.f, 0.f, 0.f, 0.f, 0.f, 0.f, 0.f};

  issue_tile(0, 0);
  for (int kt = 0; kt < Nn / 32; ++kt) {
    const int cur = kt & 1;
    const int mb = kt * 32;
    if (mb + 32 < Nn) {
      issue_tile(mb + 32, cur ^ 1);  // prefetch next tile into other buffer
      wait_async_le9();              // this tile's 9 ops done; next 9 in flight
    } else {
      wait_async_0();
    }
    __syncthreads();                 // tile visible to all 4 waves

    const u16* Kt = &KtLds[cur][0];
    const u16* Vt = &VtLds[cur][0];

    // S = Q K^T for key tile [mb, mb+32): two 16x16 WMMAs (B frags from LDS).
    v16bf kb0 = load_B32x16(Kt, LDV);
    v16bf kb1 = load_B32x16(Kt + 16 * LDV, LDV);
    v8f s0 = wmma_bf16(qf, kb0, zero8);   // cols m = mb .. mb+15
    v8f s1 = wmma_bf16(qf, kb1, zero8);   // cols m = mb+16 .. mb+31

    // Online softmax. Row r = row0+i lives in VGPR i across one 16-lane half,
    // so xor-shuffles with offsets 8..1 reduce exactly within that row.
#pragma unroll
    for (int i = 0; i < 8; ++i) {
      float mx = fmaxf(s0[i], s1[i]);
#pragma unroll
      for (int off = 8; off >= 1; off >>= 1) mx = fmaxf(mx, __shfl_xor(mx, off, 32));
      const float mnew = fmaxf(m_i[i], mx);
      const float corr = __expf(m_i[i] - mnew);
      const float p0 = __expf(s0[i] - mnew);
      const float p1 = __expf(s1[i] - mnew);
      float rs = p0 + p1;
#pragma unroll
      for (int off = 8; off >= 1; off >>= 1) rs += __shfl_xor(rs, off, 32);
      l_i[i] = l_i[i] * corr + rs;
      m_i[i] = mnew;
#pragma unroll
      for (int t = 0; t < 16; ++t) accO[t][i] *= corr;
      // Stage P (bf16) to LDS in row-major 16x32 for re-load as an A fragment.
      pw[(size_t)(row0 + i) * 32 + colc]      = f2bf(p0);
      pw[(size_t)(row0 + i) * 32 + colc + 16] = f2bf(p1);
    }
    asm volatile("s_wait_dscnt 0" ::: "memory");  // DS stores visible before DS loads
    const v16bf pa = load_A16x32(pw, 32);         // ds_load_b128 x2 per lane

    // O += P @ V^T : 16 WMMAs cover all 256 output channels (B frags from LDS).
#pragma unroll
    for (int t = 0; t < 16; ++t) {
      v16bf vbf = load_B32x16(Vt + (size_t)(t * 16) * LDV, LDV);
      accO[t] = wmma_bf16(pa, vbf, accO[t]);
    }
    __syncthreads();                 // all waves done with buf[cur] before reuse
  }

  // Epilogue: out[o, n] = scale * O[n, o] / l[n] + x[o, n]
  const float sc = scale[0];
  const float* xb = x + (size_t)b * Cc * Nn;
  float* ob = out + (size_t)b * Cc * Nn;
#pragma unroll
  for (int i = 0; i < 8; ++i) {
    const float inv_l = 1.0f / l_i[i];
    const int n = qbase + row0 + i;
#pragma unroll
    for (int t = 0; t < 16; ++t) {
      const int o = t * 16 + colc;
      const size_t idx = (size_t)o * Nn + n;
      ob[idx] = sc * (accO[t][i] * inv_l) + xb[idx];
    }
  }
}

// ---------------------------------------------------------------------------
// Host launcher.
// Workspace layout (~18.5 MB): Xh, Wqh, Wkh, Wvh, Qh, Kh, Vh (all bf16).
// ---------------------------------------------------------------------------
extern "C" void kernel_launch(void* const* d_in, const int* in_sizes, int n_in,
                              void* d_out, int out_size, void* d_ws, size_t ws_size,
                              hipStream_t stream) {
  const float* x  = (const float*)d_in[0];
  const float* Wq = (const float*)d_in[1];
  const float* Wk = (const float*)d_in[2];
  const float* Wv = (const float*)d_in[3];
  const float* sc = (const float*)d_in[4];
  float* out = (float*)d_out;

  char* ws = (char*)d_ws;
  size_t off = 0;
  auto take = [&](size_t bytes) -> char* {
    char* p = ws + off;
    off = (off + bytes + 255) & ~(size_t)255;
    return p;
  };
  u16* Xh  = (u16*)take((size_t)Bn * Nn * Cc * 2);   // x transposed, bf16 (8 MB)
  u16* Wqh = (u16*)take((size_t)Rr * Cc * 2);
  u16* Wkh = (u16*)take((size_t)Rr * Cc * 2);
  u16* Wvh = (u16*)take((size_t)Cc * Cc * 2);
  u16* Qh  = (u16*)take((size_t)Bn * Nn * Rr * 2);   // (B,N,32)
  u16* Kh  = (u16*)take((size_t)Bn * Nn * Rr * 2);   // (B,N,32)
  u16* Vh  = (u16*)take((size_t)Bn * Cc * Nn * 2);   // (B,C,N) (8 MB)

  // 0) precision conversion / transpose
  transpose_cvt_kernel<<<dim3(Nn / 32, Cc / 32, Bn), dim3(32, 8), 0, stream>>>(x, Xh);
  cvt_bf16_kernel<<<(Rr * Cc + 255) / 256, 256, 0, stream>>>(Wq, Wqh, Rr * Cc);
  cvt_bf16_kernel<<<(Rr * Cc + 255) / 256, 256, 0, stream>>>(Wk, Wkh, Rr * Cc);
  cvt_bf16_kernel<<<(Cc * Cc + 255) / 256, 256, 0, stream>>>(Wv, Wvh, Cc * Cc);

  // 1) projections (WMMA NT GEMMs)
  // Q[n,r] = sum_c Xh[n,c] * Wq[r,c]   -> 512 tiles/batch
  gemm_nt_bf16_kernel<<<dim3(128, Bn), 128, 0, stream>>>(
      Xh, (long long)Nn * Cc, Wqh, 0LL, Qh, (long long)Nn * Rr,
      Nn, Rr, Cc, Cc, Cc, Rr);
  // K[n,r] = sum_c Xh[n,c] * Wk[r,c]
  gemm_nt_bf16_kernel<<<dim3(128, Bn), 128, 0, stream>>>(
      Xh, (long long)Nn * Cc, Wkh, 0LL, Kh, (long long)Nn * Rr,
      Nn, Rr, Cc, Cc, Cc, Rr);
  // V[o,m] = sum_c Wv[o,c] * Xh[m,c]   -> 4096 tiles/batch
  gemm_nt_bf16_kernel<<<dim3(1024, Bn), 128, 0, stream>>>(
      Wvh, 0LL, Xh, (long long)Nn * Cc, Vh, (long long)Cc * Nn,
      Cc, Nn, Cc, Cc, Cc, Nn);

  // 2) fused flash attention + scale*out + x
  flash_attn_kernel<<<dim3(Nn / 64, Bn), 128, 0, stream>>>(Qh, Kh, Vh, x, sc, out);
}
